// TransformerEncoderLayer_62139586839344
// MI455X (gfx1250) — matrix operand order repaired and checked by reading, not compiled, not run on hardware
//
#include <hip/hip_runtime.h>
#include <stdint.h>

typedef __attribute__((ext_vector_type(16))) __bf16 v16bf;
typedef __attribute__((ext_vector_type(8)))  float  v8f;

#define D_MODEL 1024
#define SEQ     2048
#define BATCH   2
#define NHEAD   16
#define HDIM    64
#define DFF     4096
#define ROWS    (BATCH*SEQ)   // 4096
#define LN_EPS  1e-5f

__device__ __forceinline__ unsigned short f2bf(float f) {
    unsigned u = __float_as_uint(f);
    u += 0x7FFFu + ((u >> 16) & 1u);     // round-to-nearest-even
    return (unsigned short)(u >> 16);
}

union BfFrag { unsigned u[8]; v16bf v; };

// ---- CDNA5 async global->LDS copy (ASYNCcnt-tracked, per-lane addressed) ----
__device__ __forceinline__ void async_b128(unsigned lds_byte, const void* gptr) {
    asm volatile("global_load_async_to_lds_b128 %0, %1, off"
                 :: "v"(lds_byte), "v"(gptr) : "memory");
}
__device__ __forceinline__ void wait_async0()  { asm volatile("s_wait_asynccnt 0x0" ::: "memory"); }
__device__ __forceinline__ void wait_async4()  { asm volatile("s_wait_asynccnt 0x4" ::: "memory"); }
__device__ __forceinline__ void wait_async12() { asm volatile("s_wait_asynccnt 0xc" ::: "memory"); }
__device__ __forceinline__ unsigned lds_addr(const void* p) {
    return (unsigned)(uintptr_t)p;   // flat LDS address truncates to DS offset (ISA 10.2)
}

// ---------------------------------------------------------------- cvt f32->bf16
__global__ __launch_bounds__(256) void cvt_bf16_kernel(const float* __restrict__ in,
                                                       unsigned short* __restrict__ out,
                                                       int n) {
    int i = blockIdx.x * blockDim.x + threadIdx.x;
    int stride = gridDim.x * blockDim.x;
    for (; i < n; i += stride) out[i] = f2bf(in[i]);
}

// ---------------------------------------------------------------- GEMM: Y = act(A[M,K] @ W[N,K]^T + bias)
// 128x64 block tile, K-step 64, 128 threads = 4 wave32; each wave computes 32 rows x 64 cols
// (2 row-tiles x 4 col-tiles = 8 accumulators, 16 WMMA per staged tile).
// Double-buffered async (global -> LDS) staging: 12 async ops/wave/stage.
__global__ __launch_bounds__(128) void gemm_bf16_wmma(
    const unsigned short* __restrict__ A,
    const unsigned short* __restrict__ W,
    const float* __restrict__ bias,
    unsigned short* __restrict__ outB,
    float* __restrict__ outF,
    int M, int N, int K, int relu)
{
    __shared__ __align__(16) unsigned short As[2][128 * 64];  // 32 KB
    __shared__ __align__(16) unsigned short Ws[2][64 * 64];   // 16 KB
    const int tid  = threadIdx.x;
    const int lane = tid & 31;
    const int wave = tid >> 5;
    const int half = lane >> 4;
    const int l16  = lane & 15;
    const int gm0  = blockIdx.y * 128;
    const int gn0  = blockIdx.x * 64;
    const unsigned asBase = lds_addr(&As[0][0]);
    const unsigned wsBase = lds_addr(&Ws[0][0]);

    v8f zero = {0.f,0.f,0.f,0.f,0.f,0.f,0.f,0.f};
    v8f acc[2][4];
#pragma unroll
    for (int mt = 0; mt < 2; ++mt)
#pragma unroll
        for (int t = 0; t < 4; ++t) acc[mt][t] = zero;

    // stage one K-slab: A 128x64 (1024 b128 chunks) + W 64x64 (512 chunks)
    auto stage = [&](int buf, int k0) {
        const unsigned aB = asBase + buf * 16384;
        const unsigned wB = wsBase + buf * 8192;
#pragma unroll
        for (int c0 = 0; c0 < 1024; c0 += 128) {
            int c = c0 + tid;
            int row = c >> 3, q = c & 7;
            async_b128(aB + c * 16, &A[(size_t)(gm0 + row) * K + k0 + q * 8]);
        }
#pragma unroll
        for (int c0 = 0; c0 < 512; c0 += 128) {
            int c = c0 + tid;
            int row = c >> 3, q = c & 7;
            async_b128(wB + c * 16, &W[(size_t)(gn0 + row) * K + k0 + q * 8]);
        }
    };

    const int nk = K >> 6;
    stage(0, 0);
    for (int it = 0; it < nk; ++it) {
        const int cur = it & 1;
        if (it + 1 < nk) {
            stage(cur ^ 1, (it + 1) << 6);
            wait_async12();          // previous stage's 12 ops (in-order) are done
        } else {
            wait_async0();
        }
        __syncthreads();

        const unsigned short* Ab = As[cur];
        const unsigned short* Wb = Ws[cur];
#pragma unroll
        for (int kc = 0; kc < 2; ++kc) {
            // load all fragments for this k-chunk first (independent ds_loads)
            BfFrag af[2];
#pragma unroll
            for (int mt = 0; mt < 2; ++mt) {
                const int arow = wave * 32 + mt * 16 + l16;
#pragma unroll
                for (int v = 0; v < 8; ++v) {
                    // A fragment 16x32: K = half*16 + 2v (contiguous, matches B)
                    int kk = kc * 32 + half * 16 + 2 * v;
                    af[mt].u[v] = *(const unsigned*)&Ab[arow * 64 + kk];
                }
            }
            BfFrag bf[4];
#pragma unroll
            for (int t = 0; t < 4; ++t) {
                // B fragment 32x16: lane col = l16, K = half*16 + 2v (contiguous in W row)
                const int ncol = t * 16 + l16;
#pragma unroll
                for (int v = 0; v < 8; ++v)
                    bf[t].u[v] = *(const unsigned*)&Wb[ncol * 64 + kc * 32 + half * 16 + 2 * v];
            }
#pragma unroll
            for (int mt = 0; mt < 2; ++mt)
#pragma unroll
                for (int t = 0; t < 4; ++t)
                    acc[mt][t] = __builtin_amdgcn_wmma_f32_16x16x32_bf16(
                        false, af[mt].v, false, bf[t].v, (short)0, acc[mt][t], false, false);
        }
        __syncthreads();
    }

    // epilogue: C layout — VGPR r holds row half*8+r, col = l16 (per 16x16 tile)
#pragma unroll
    for (int mt = 0; mt < 2; ++mt)
#pragma unroll
        for (int t = 0; t < 4; ++t) {
            const int gn = gn0 + t * 16 + l16;
            const float bv = bias ? bias[gn] : 0.f;
#pragma unroll
            for (int r = 0; r < 8; ++r) {
                const int gm = gm0 + wave * 32 + mt * 16 + half * 8 + r;
                float y = acc[mt][t][r] + bv;
                if (relu) y = y > 0.f ? y : 0.f;
                if (outF) outF[(size_t)gm * N + gn] = y;
                if (outB) outB[(size_t)gm * N + gn] = f2bf(y);
            }
        }
}

// ---------------------------------------------------------------- flash attention
// grid.x = B*H, grid.y = SEQ/64. Block = 128 (4 waves), wave owns 16 Q rows.
// K/V tiles double-buffered via async LDS loads (4 ops/wave/stage).
__global__ __launch_bounds__(128) void attn_flash_wmma(
    const unsigned short* __restrict__ QKV,   // [ROWS, 3*D_MODEL], cols = sel*1024 + h*64 + d
    unsigned short* __restrict__ ctx)         // [ROWS, D_MODEL]
{
    __shared__ __align__(16) unsigned short Ks[2][32 * 64];
    __shared__ __align__(16) unsigned short Vs[2][32 * 64];
    __shared__ __align__(16) unsigned short Ps[4][16 * 32];
    const int tid  = threadIdx.x;
    const int lane = tid & 31;
    const int wave = tid >> 5;
    const int half = lane >> 4;
    const int l16  = lane & 15;
    const int b    = blockIdx.x >> 4;
    const int h    = blockIdx.x & 15;
    const int q0   = blockIdx.y * 64 + wave * 16;
    const size_t rs = 3 * D_MODEL;
    const unsigned short* Qb = QKV + (size_t)(b * SEQ) * rs + h * HDIM;
    const unsigned short* Kb = Qb + D_MODEL;
    const unsigned short* Vb = Qb + 2 * D_MODEL;
    const float scale = 0.125f;  // 1/sqrt(64)
    const unsigned ksBase = lds_addr(&Ks[0][0]);
    const unsigned vsBase = lds_addr(&Vs[0][0]);

    // Q strip (16 x 64) as two 16x32 A-fragments, loaded straight from global
    BfFrag aq[2];
#pragma unroll
    for (int kc = 0; kc < 2; ++kc)
#pragma unroll
        for (int v = 0; v < 8; ++v) {
            int kk = kc * 32 + half * 16 + 2 * v;
            aq[kc].u[v] = *(const unsigned*)&Qb[(size_t)(q0 + l16) * rs + kk];
        }

    v8f zero = {0.f,0.f,0.f,0.f,0.f,0.f,0.f,0.f};
    v8f accO[4];
#pragma unroll
    for (int t = 0; t < 4; ++t) accO[t] = zero;
    float mrow[8], lrow[8];
#pragma unroll
    for (int r = 0; r < 8; ++r) { mrow[r] = -__builtin_inff(); lrow[r] = 0.f; }

    auto stage = [&](int buf, int m0) {
#pragma unroll
        for (int c0 = 0; c0 < 256; c0 += 128) {
            int c = c0 + tid;
            int row = c >> 3, q = c & 7;
            async_b128(ksBase + buf * 4096 + c * 16, &Kb[(size_t)(m0 + row) * rs + q * 8]);
            async_b128(vsBase + buf * 4096 + c * 16, &Vb[(size_t)(m0 + row) * rs + q * 8]);
        }
    };

    const int niter = SEQ / 32;
    stage(0, 0);
    for (int it = 0; it < niter; ++it) {
        const int cur = it & 1;
        if (it + 1 < niter) {
            stage(cur ^ 1, (it + 1) * 32);
            wait_async4();
        } else {
            wait_async0();
        }
        __syncthreads();
        const unsigned short* Kt = Ks[cur];
        const unsigned short* Vt = Vs[cur];

        // S (16 x 32) = Q @ K^T : two 16x16 tiles, each over two K chunks
        v8f accS[2];
#pragma unroll
        for (int st = 0; st < 2; ++st) {
            accS[st] = zero;
#pragma unroll
            for (int kc = 0; kc < 2; ++kc) {
                BfFrag bk;  // B col n = key row (st*16+l16); K(d) = kc*32 + half*16 + 2v
#pragma unroll
                for (int v = 0; v < 8; ++v)
                    bk.u[v] = *(const unsigned*)&Kt[(st * 16 + l16) * 64 + kc * 32 + half * 16 + 2 * v];
                accS[st] = __builtin_amdgcn_wmma_f32_16x16x32_bf16(
                    false, aq[kc].v, false, bk.v, (short)0, accS[st], false, false);
            }
        }

        // online softmax, rows half*8+r; reductions across the 16 lanes of this half
#pragma unroll
        for (int r = 0; r < 8; ++r) {
            float s0 = accS[0][r] * scale, s1 = accS[1][r] * scale;
            float rm = fmaxf(s0, s1);
            rm = fmaxf(rm, __shfl_xor(rm, 1));
            rm = fmaxf(rm, __shfl_xor(rm, 2));
            rm = fmaxf(rm, __shfl_xor(rm, 4));
            rm = fmaxf(rm, __shfl_xor(rm, 8));
            float mnew = fmaxf(mrow[r], rm);
            float corr = __expf(mrow[r] - mnew);
            float p0 = __expf(s0 - mnew), p1 = __expf(s1 - mnew);
            float rsum = p0 + p1;
            rsum += __shfl_xor(rsum, 1);
            rsum += __shfl_xor(rsum, 2);
            rsum += __shfl_xor(rsum, 4);
            rsum += __shfl_xor(rsum, 8);
            lrow[r] = lrow[r] * corr + rsum;
            mrow[r] = mnew;
#pragma unroll
            for (int t = 0; t < 4; ++t) accO[t][r] *= corr;
            Ps[wave][(half * 8 + r) * 32 + l16]      = f2bf(p0);
            Ps[wave][(half * 8 + r) * 32 + 16 + l16] = f2bf(p1);
        }

        // O (16 x 64) += P(16x32) @ V(32x64)
        BfFrag ap;
#pragma unroll
        for (int v = 0; v < 8; ++v) {
            int kk = half * 16 + 2 * v;
            ap.u[v] = *(const unsigned*)&Ps[wave][l16 * 32 + kk];
        }
#pragma unroll
        for (int t = 0; t < 4; ++t) {
            BfFrag bv;  // B col n = d within tile; V stored [k][d], pairs strided by 64
#pragma unroll
            for (int v = 0; v < 8; ++v) {
                int kk = half * 16 + 2 * v;
                unsigned lo = Vt[kk * 64 + t * 16 + l16];
                unsigned hi = Vt[(kk + 1) * 64 + t * 16 + l16];
                bv.u[v] = lo | (hi << 16);
            }
            accO[t] = __builtin_amdgcn_wmma_f32_16x16x32_bf16(
                false, ap.v, false, bv.v, (short)0, accO[t], false, false);
        }
        __syncthreads();
    }

    // normalize and write context in [B*N, C] bf16, cols h*64 + d
#pragma unroll
    for (int r = 0; r < 8; ++r) {
        const float inv = 1.f / lrow[r];
        const int gm = b * SEQ + q0 + half * 8 + r;
#pragma unroll
        for (int t = 0; t < 4; ++t)
            ctx[(size_t)gm * D_MODEL + h * HDIM + t * 16 + l16] = f2bf(accO[t][r] * inv);
    }
}

// ---------------------------------------------------------------- add + layernorm (row per block)
__global__ __launch_bounds__(256) void add_ln_kernel(
    const float* __restrict__ a, const float* __restrict__ bsrc,
    const float* __restrict__ gamma, const float* __restrict__ beta,
    float* __restrict__ outF, unsigned short* __restrict__ outB)
{
    __shared__ float red[256];
    __shared__ float red2[256];
    const int row = blockIdx.x;
    const int tid = threadIdx.x;
    float x[4];
    float s = 0.f, sq = 0.f;
#pragma unroll
    for (int i = 0; i < 4; ++i) {
        int c = tid * 4 + i;
        float v = a[(size_t)row * D_MODEL + c] + bsrc[(size_t)row * D_MODEL + c];
        x[i] = v; s += v; sq += v * v;
    }
    red[tid] = s; red2[tid] = sq;
    __syncthreads();
    for (int off = 128; off > 0; off >>= 1) {
        if (tid < off) { red[tid] += red[tid + off]; red2[tid] += red2[tid + off]; }
        __syncthreads();
    }
    const float mu   = red[0] * (1.f / D_MODEL);
    const float var  = red2[0] * (1.f / D_MODEL) - mu * mu;
    const float rstd = rsqrtf(var + LN_EPS);
#pragma unroll
    for (int i = 0; i < 4; ++i) {
        int c = tid * 4 + i;
        float y = (x[i] - mu) * rstd * gamma[c] + beta[c];
        outF[(size_t)row * D_MODEL + c] = y;
        if (outB) outB[(size_t)row * D_MODEL + c] = f2bf(y);
    }
}

// ---------------------------------------------------------------- driver
extern "C" void kernel_launch(void* const* d_in, const int* in_sizes, int n_in,
                              void* d_out, int out_size, void* d_ws, size_t ws_size,
                              hipStream_t stream)
{
    const float* src    = (const float*)d_in[0];
    const float* w_qkv  = (const float*)d_in[1];
    const float* w_proj = (const float*)d_in[2];
    const float* b_proj = (const float*)d_in[3];
    const float* w1     = (const float*)d_in[4];
    const float* b1     = (const float*)d_in[5];
    const float* w2     = (const float*)d_in[6];
    const float* b2     = (const float*)d_in[7];
    const float* g1     = (const float*)d_in[8];
    const float* be1    = (const float*)d_in[9];
    const float* g2     = (const float*)d_in[10];
    const float* be2    = (const float*)d_in[11];
    float* out = (float*)d_out;

    char* ws = (char*)d_ws;
    size_t off = 0;
    auto alloc = [&](size_t bytes) -> char* {
        char* p = ws + off;
        off += (bytes + 255) & ~(size_t)255;
        return p;
    };
    unsigned short* wqkv_bf  = (unsigned short*)alloc((size_t)3 * D_MODEL * D_MODEL * 2); //  6 MB
    unsigned short* wproj_bf = (unsigned short*)alloc((size_t)D_MODEL * D_MODEL * 2);     //  2 MB
    unsigned short* w1_bf    = (unsigned short*)alloc((size_t)DFF * D_MODEL * 2);         //  8 MB
    unsigned short* w2_bf    = (unsigned short*)alloc((size_t)D_MODEL * DFF * 2);         //  8 MB
    unsigned short* x_bf     = (unsigned short*)alloc((size_t)ROWS * D_MODEL * 2);        //  8 MB (src bf16, later LN1 bf16)
    unsigned short* big      = (unsigned short*)alloc((size_t)ROWS * DFF * 2);            // 32 MB (QKV bf16, later FFN hidden)
    unsigned short* ctx_bf   = (unsigned short*)alloc((size_t)ROWS * D_MODEL * 2);        //  8 MB
    float*          f32scr   = (float*)alloc((size_t)ROWS * D_MODEL * 4);                 // 16 MB (proj out, later ffn out)
    float*          x_f32    = (float*)alloc((size_t)ROWS * D_MODEL * 4);                 // 16 MB
    unsigned short* qkv_bf = big;   // ROWS x 3072 (25.2 MB) fits in 32 MB region
    unsigned short* h_bf   = big;   // ROWS x 4096, used after attention is done

    // 1) fp32 -> bf16 conversions
    cvt_bf16_kernel<<<2048, 256, 0, stream>>>(src,    x_bf,     ROWS * D_MODEL);
    cvt_bf16_kernel<<<2048, 256, 0, stream>>>(w_qkv,  wqkv_bf,  3 * D_MODEL * D_MODEL);
    cvt_bf16_kernel<<<2048, 256, 0, stream>>>(w_proj, wproj_bf, D_MODEL * D_MODEL);
    cvt_bf16_kernel<<<2048, 256, 0, stream>>>(w1,     w1_bf,    DFF * D_MODEL);
    cvt_bf16_kernel<<<2048, 256, 0, stream>>>(w2,     w2_bf,    D_MODEL * DFF);

    const dim3 blk(128);
    // 2) QKV = src @ w_qkv^T   [4096 x 3072]
    gemm_bf16_wmma<<<dim3(3 * D_MODEL / 64, ROWS / 128), blk, 0, stream>>>(
        x_bf, wqkv_bf, nullptr, qkv_bf, nullptr, ROWS, 3 * D_MODEL, D_MODEL, 0);
    // 3) flash attention -> ctx bf16
    attn_flash_wmma<<<dim3(BATCH * NHEAD, SEQ / 64), blk, 0, stream>>>(qkv_bf, ctx_bf);
    // 4) proj = ctx @ w_proj^T + b_proj  (f32)
    gemm_bf16_wmma<<<dim3(D_MODEL / 64, ROWS / 128), blk, 0, stream>>>(
        ctx_bf, wproj_bf, b_proj, nullptr, f32scr, ROWS, D_MODEL, D_MODEL, 0);
    // 5) x = LN(src + proj); also bf16 copy for FFN
    add_ln_kernel<<<ROWS, 256, 0, stream>>>(src, f32scr, g1, be1, x_f32, x_bf);
    // 6) h = relu(x @ w1^T + b1)  (bf16)
    gemm_bf16_wmma<<<dim3(DFF / 64, ROWS / 128), blk, 0, stream>>>(
        x_bf, w1_bf, b1, h_bf, nullptr, ROWS, DFF, D_MODEL, 1);
    // 7) ffn = h @ w2^T + b2  (f32)
    gemm_bf16_wmma<<<dim3(D_MODEL / 64, ROWS / 128), blk, 0, stream>>>(
        h_bf, w2_bf, b2, nullptr, f32scr, ROWS, D_MODEL, DFF, 0);
    // 8) out = LN(x + ffn)
    add_ln_kernel<<<ROWS, 256, 0, stream>>>(x_f32, f32scr, g2, be2, out, nullptr);
}
